// DoubleAttention_14078902796707
// MI455X (gfx1250) — matrix operand order repaired and checked by reading, not compiled, not run on hardware
//
#include <hip/hip_runtime.h>
#include <stdint.h>

// ---------- types for WMMA ----------
typedef __attribute__((ext_vector_type(8)))  __bf16 v8bf;
typedef __attribute__((ext_vector_type(16))) __bf16 v16bf;
typedef __attribute__((ext_vector_type(8)))  float  v8f;

// ---------- helpers ----------
__device__ __forceinline__ uint16_t f32_to_bf16_rne(float x) {
    uint32_t u = __float_as_uint(x);
    u += 0x7fffu + ((u >> 16) & 1u);
    return (uint16_t)(u >> 16);
}
__device__ __forceinline__ float bf16_to_f32(uint16_t h) {
    return __uint_as_float(((uint32_t)h) << 16);
}

// 64B (4 x b128) async global -> LDS fill, ASYNCcnt-tracked.
// INST_OFFSET is added to both the LDS and global addresses (ISA 10.7/15.18),
// so one base pair covers the whole 64B chunk.
__device__ __forceinline__ void async_fill_64B(uint32_t lds_off, const uint16_t* gp) {
    uint64_t ga = (uint64_t)(uintptr_t)gp;
    asm volatile(
        "global_load_async_to_lds_b128 %0, %1, off\n\t"
        "global_load_async_to_lds_b128 %0, %1, off offset:16\n\t"
        "global_load_async_to_lds_b128 %0, %1, off offset:32\n\t"
        "global_load_async_to_lds_b128 %0, %1, off offset:48"
        :: "v"(lds_off), "v"(ga) : "memory");
}
__device__ __forceinline__ void wait_async0() {
    asm volatile("s_wait_asynccnt 0x0" ::: "memory");
}
__device__ __forceinline__ uint32_t lds_addr(const void* p) {
    // LDS generic address: low 32 bits are the LDS offset (aperture in high bits).
    return (uint32_t)(uintptr_t)p;
}

// ---------- f32 -> bf16 convert (grid-stride) ----------
__global__ __launch_bounds__(256) void k_f32_to_bf16(const float* __restrict__ in,
                                                     uint16_t* __restrict__ out, int n) {
    for (int i = blockIdx.x * blockDim.x + threadIdx.x; i < n; i += gridDim.x * blockDim.x)
        out[i] = f32_to_bf16_rne(in[i]);
}

// ---------- in-place row softmax over bf16 rows of length 1024 ----------
__global__ __launch_bounds__(256) void k_softmax_row1024(uint16_t* __restrict__ data) {
    const int row = blockIdx.x;
    const int t = threadIdx.x;
    uint16_t* p = data + (size_t)row * 1024;
    __shared__ float red[256];

    float v[4];
    float m = -3.402823e38f;
#pragma unroll
    for (int i = 0; i < 4; ++i) {
        v[i] = bf16_to_f32(p[t + i * 256]);
        m = fmaxf(m, v[i]);
    }
    red[t] = m; __syncthreads();
    for (int s = 128; s > 0; s >>= 1) {
        if (t < s) red[t] = fmaxf(red[t], red[t + s]);
        __syncthreads();
    }
    m = red[0]; __syncthreads();

    float sum = 0.f;
#pragma unroll
    for (int i = 0; i < 4; ++i) { v[i] = __expf(v[i] - m); sum += v[i]; }
    red[t] = sum; __syncthreads();
    for (int s = 128; s > 0; s >>= 1) {
        if (t < s) red[t] += red[t + s];
        __syncthreads();
    }
    const float inv = 1.0f / red[0];
#pragma unroll
    for (int i = 0; i < 4; ++i) p[t + i * 256] = f32_to_bf16_rne(v[i] * inv);
}

// ---------- tiled WMMA GEMM, async-to-LDS, double-buffered, 1 barrier/iter ----------
// C[M,Nn] = A[M,K](row-major bf16) x B (+ bias[m])
//   BT=false: B is [K,Nn] row-major (async fill).
//   BT=true : B is [Nn,K] row-major (C = A*B^T); B register-staged + transposed ds_store.
// Block: 256 threads (8 waves). Tile 128x128, BK=64 (two WMMA K-slices per step).
// Wave grid 4(M) x 2(N): each wave -> 32x64 = 2x4 accumulators of 16x16.
#define BM 128
#define BN 128
#define BK 64
#define LDAP 72    // padded LDS stride (bf16), 144B rows (16B multiple)
#define LDBP 136   // padded LDS stride (bf16), 272B rows (16B multiple)

template <bool BT, bool BIAS, bool OBF16>
__global__ __launch_bounds__(256) void k_gemm_wmma(
    const uint16_t* __restrict__ A, long long strideA,
    const uint16_t* __restrict__ Bm, long long strideB,
    const float* __restrict__ bias,
    void* __restrict__ Cptr, long long strideC,
    int M, int Nn, int K)
{
    A  += (size_t)blockIdx.z * strideA;
    Bm += (size_t)blockIdx.z * strideB;

    const int m0 = blockIdx.y * BM;
    const int n0 = blockIdx.x * BN;
    const int tid  = threadIdx.x;
    const int lane = tid & 31;
    const int wid  = tid >> 5;
    const int wm   = wid & 3;     // 0..3 -> M
    const int wn   = wid >> 2;    // 0..1 -> N
    const int half = lane >> 4;
    const int r    = lane & 15;

    __shared__ __align__(16) uint16_t sA[2 * BM * LDAP];
    __shared__ __align__(16) uint16_t sB[2 * BK * LDBP];

    // per-thread fill geometry
    const int a_row = tid >> 1, a_ko = (tid & 1) * 32;             // A: 2 thr/row, 64B each
    const int b_row = tid >> 2, b_no = (tid & 3) * 32;             // B NN: 4 thr/row, 64B each
    const int t_n   = tid >> 1, t_ko = (tid & 1) * 32;             // B NT: 2 thr/n-row, 64B each

    v8f acc[2][4];
#pragma unroll
    for (int mi = 0; mi < 2; ++mi)
#pragma unroll
        for (int ni = 0; ni < 4; ++ni) acc[mi][ni] = (v8f){};

    uint4 breg[4];  // NT staging regs (32 bf16 along K)

    auto issue_A = [&](int k0, int buf) {
        async_fill_64B(lds_addr(&sA[buf * BM * LDAP + a_row * LDAP + a_ko]),
                       A + (size_t)(m0 + a_row) * K + k0 + a_ko);
    };
    auto issue_B_nn = [&](int k0, int buf) {
        async_fill_64B(lds_addr(&sB[buf * BK * LDBP + b_row * LDBP + b_no]),
                       Bm + (size_t)(k0 + b_row) * Nn + n0 + b_no);
    };
    auto load_B_nt = [&](int k0) {
        const uint4* g = (const uint4*)(Bm + (size_t)(n0 + t_n) * K + k0 + t_ko);
#pragma unroll
        for (int j = 0; j < 4; ++j) breg[j] = g[j];
    };
    auto store_B_nt = [&](int buf) {
        uint16_t* s = &sB[buf * BK * LDBP];
#pragma unroll
        for (int j = 0; j < 4; ++j) {
            uint32_t w[4] = {breg[j].x, breg[j].y, breg[j].z, breg[j].w};
#pragma unroll
            for (int e = 0; e < 4; ++e) {
                s[(t_ko + j * 8 + 2 * e + 0) * LDBP + t_n] = (uint16_t)(w[e] & 0xffffu);
                s[(t_ko + j * 8 + 2 * e + 1) * LDBP + t_n] = (uint16_t)(w[e] >> 16);
            }
        }
    };

    // prologue
    issue_A(0, 0);
    if (BT) load_B_nt(0); else issue_B_nn(0, 0);

    for (int k0 = 0, it = 0; k0 < K; k0 += BK, ++it) {
        const int cur = it & 1;
        if (BT) store_B_nt(cur);
        wait_async0();
        __syncthreads();

        const int kn = k0 + BK;
        if (kn < K) {
            issue_A(kn, cur ^ 1);
            if (BT) load_B_nt(kn); else issue_B_nn(kn, cur ^ 1);
        }

        const uint16_t* sAb = &sA[cur * BM * LDAP];
        const uint16_t* sBb = &sB[cur * BK * LDBP];
#pragma unroll
        for (int kk = 0; kk < BK; kk += 32) {
            // A (16x32): lane half 0 -> K 0-7 & 16-23; half 1 -> K 8-15 & 24-31; row = M
            v16bf afrag[2];
#pragma unroll
            for (int mi = 0; mi < 2; ++mi) {
                const uint16_t* base = &sAb[(wm * 32 + mi * 16 + r) * LDAP + kk + half * 8];
                v8bf lo = *(const v8bf*)(base);
                v8bf hi = *(const v8bf*)(base + 16);
                afrag[mi] = __builtin_shufflevector(lo, hi,
                    0,1,2,3,4,5,6,7,8,9,10,11,12,13,14,15);
            }
            // B (32x16): lane half 0 -> K 0-7 & 16-23; half 1 -> K 8-15 & 24-31; col = N
            v16bf bfrag[4];
#pragma unroll
            for (int ni = 0; ni < 4; ++ni) {
                const uint16_t* base = &sBb[(kk + half * 8) * LDBP + wn * 64 + ni * 16 + r];
#pragma unroll
                for (int s = 0; s < 8; ++s) {
                    bfrag[ni][s]     = __builtin_bit_cast(__bf16, base[s * LDBP]);
                    bfrag[ni][s + 8] = __builtin_bit_cast(__bf16, base[(s + 16) * LDBP]);
                }
            }
#pragma unroll
            for (int mi = 0; mi < 2; ++mi)
#pragma unroll
                for (int ni = 0; ni < 4; ++ni)
                    acc[mi][ni] = __builtin_amdgcn_wmma_f32_16x16x32_bf16(
                        false, afrag[mi], false, bfrag[ni],
                        (short)0, acc[mi][ni], false, false);
        }
    }

    // ---- epilogue: C layout VGPR v -> M = v + 8*half, N = lane&15 ----
#pragma unroll
    for (int mi = 0; mi < 2; ++mi) {
#pragma unroll
        for (int ni = 0; ni < 4; ++ni) {
            const int col = n0 + wn * 64 + ni * 16 + r;
            const int rbase = m0 + wm * 32 + mi * 16 + half * 8;
#pragma unroll
            for (int v = 0; v < 8; ++v) {
                const int row = rbase + v;
                float val = acc[mi][ni][v];
                if (BIAS) val += bias[row];
                const size_t o = (size_t)blockIdx.z * strideC + (size_t)row * Nn + col;
                if (OBF16) ((uint16_t*)Cptr)[o] = f32_to_bf16_rne(val);
                else       ((float*)Cptr)[o]   = val;
            }
        }
    }
}

// ---------- host-side launch ----------
extern "C" void kernel_launch(void* const* d_in, const int* in_sizes, int n_in,
                              void* d_out, int out_size, void* d_ws, size_t ws_size,
                              hipStream_t stream) {
    const float* x  = (const float*)d_in[0];
    const float* wA = (const float*)d_in[1];
    const float* bA = (const float*)d_in[2];
    const float* wB = (const float*)d_in[3];
    const float* bB = (const float*)d_in[4];
    const float* wV = (const float*)d_in[5];
    const float* bV = (const float*)d_in[6];
    const float* wR = (const float*)d_in[7];
    const float* bR = (const float*)d_in[8];
    float* out = (float*)d_out;

    const int Bsz = 32, C = 512, HW = 1024, Co = 512, dn = 512;
    const size_t NX = (size_t)Bsz * C * HW;   // 16,777,216
    const size_t NW = (size_t)Co * C;         // 262,144
    const size_t NG = (size_t)Bsz * Co * dn;  // 8,388,608

    char* ws = (char*)d_ws;
    size_t off = 0;
    auto alloc = [&](size_t bytes) -> void* {
        void* p = ws + off;
        off += (bytes + 255) & ~(size_t)255;
        return p;
    };
    uint16_t* xb   = (uint16_t*)alloc(NX * 2);
    uint16_t* wAb  = (uint16_t*)alloc(NW * 2);
    uint16_t* wBb  = (uint16_t*)alloc(NW * 2);
    uint16_t* wVb  = (uint16_t*)alloc(NW * 2);
    uint16_t* wRb  = (uint16_t*)alloc(NW * 2);
    uint16_t* Abuf = (uint16_t*)alloc(NX * 2);
    uint16_t* Mbuf = (uint16_t*)alloc(NX * 2);  // Bm pre-act, softmax in place
    uint16_t* Vbuf = (uint16_t*)alloc(NX * 2);  // Vm pre-act, softmax in place
    uint16_t* Gbuf = (uint16_t*)alloc(NG * 2);  // glob_desc
    uint16_t* Pbuf = (uint16_t*)alloc(NG * 2);  // wR @ G
    (void)ws_size;

    // stage 0: f32 -> bf16
    k_f32_to_bf16<<<4096, 256, 0, stream>>>(x,  xb,  (int)NX);
    k_f32_to_bf16<<<256,  256, 0, stream>>>(wA, wAb, (int)NW);
    k_f32_to_bf16<<<256,  256, 0, stream>>>(wB, wBb, (int)NW);
    k_f32_to_bf16<<<256,  256, 0, stream>>>(wV, wVb, (int)NW);
    k_f32_to_bf16<<<256,  256, 0, stream>>>(wR, wRb, (int)NW);

    const long long sX = (long long)C * HW;   // per-batch x stride
    const long long sO = (long long)Co * HW;  // per-batch [512,1024] stride
    const long long sG = (long long)Co * dn;  // per-batch [512,512] stride

    // stage 1: A / Bm / Vm 1x1 convs  (W[512,512] x x_b[512,1024] + bias)
    dim3 g1(HW / BN, Co / BM, Bsz);
    k_gemm_wmma<false, true, true><<<g1, 256, 0, stream>>>(wAb, 0, xb, sX, bA, Abuf, sO, Co, HW, C);
    k_gemm_wmma<false, true, true><<<g1, 256, 0, stream>>>(wBb, 0, xb, sX, bB, Mbuf, sO, Co, HW, C);
    k_gemm_wmma<false, true, true><<<g1, 256, 0, stream>>>(wVb, 0, xb, sX, bV, Vbuf, sO, Co, HW, C);

    // stage 2: softmax over N=1024, in place
    k_softmax_row1024<<<Bsz * dn, 256, 0, stream>>>(Mbuf);
    k_softmax_row1024<<<Bsz * dn, 256, 0, stream>>>(Vbuf);

    // stage 3: G_b = A_b[512,1024] @ attnM_b[512,1024]^T  (NT, K=1024)
    dim3 g3(dn / BN, Co / BM, Bsz);
    k_gemm_wmma<true, false, true><<<g3, 256, 0, stream>>>(Abuf, sO, Mbuf, sO, nullptr, Gbuf, sG, Co, dn, HW);

    // stage 4: P_b = wR[512,512] @ G_b[512,512]  (NN, K=512)
    dim3 g4(dn / BN, Co / BM, Bsz);
    k_gemm_wmma<false, false, true><<<g4, 256, 0, stream>>>(wRb, 0, Gbuf, sG, nullptr, Pbuf, sG, Co, dn, Co);

    // stage 5: out_b = P_b[512,512] @ attnV_b[512,1024] + bR  (NN, K=512, f32 out)
    dim3 g5(HW / BN, Co / BM, Bsz);
    k_gemm_wmma<false, true, false><<<g5, 256, 0, stream>>>(Pbuf, sG, Vbuf, sO, bR, out, sO, Co, HW, dn);
}
